// MambaVisionMixer_18391049962109
// MI455X (gfx1250) — matrix operand order repaired and checked by reading, not compiled, not run on hardware
//
#include <hip/hip_runtime.h>
#include <hip/hip_bf16.h>

// ---------------------------------------------------------------------------
// MambaVisionMixer forward for MI455X (gfx1250).
//  * GEMMs: v_wmma_f32_16x16x32_bf16, LDS-staged with register double
//    buffering (global loads for tile k+1 in flight during WMMA on tile k).
//  * Selective scan: chunked 3-pass formulation (local scan / carry scan /
//    fixup) for 16x more parallelism than the naive serial scan.
// ---------------------------------------------------------------------------

typedef __attribute__((ext_vector_type(16))) __bf16 v16bf;
typedef __attribute__((ext_vector_type(8)))  __bf16 v8bf;
typedef __attribute__((ext_vector_type(4)))  __bf16 v4bf;
typedef __attribute__((ext_vector_type(8)))  float  v8f;

#define BSZ    8
#define SEQ    1024
#define DMODEL 768
#define DINNER 1536
#define DHALF  768
#define DSTATE 16
#define DTRANK 48
#define ROWS   (BSZ * SEQ)   // 8192
#define CHUNK  64
#define NCH    (SEQ / CHUNK) // 16

__device__ __forceinline__ float softplus_f(float x) {
    return (x > 20.0f) ? x : __logf(1.0f + __expf(x));
}

// ---- tiled WMMA GEMM: C[M,N] = A[M,K(lda)] * B[N,K(ldb)]^T (+ bias[n]) ----
#define TM 128
#define TN 64
#define TK 32
#define LSTR 40   // bf16 elems per LDS row (32 + 8 pad) -> 80B, 16B-aligned

struct StageRegs { float4 a[4]; float4 b[2]; };

// Issue all global loads for one (A,B) k-tile into registers. Addresses are
// clamped into the allocation so no branching / exec masking is needed.
__device__ __forceinline__ void stage_issue(
    const float* __restrict__ A, const float* __restrict__ Bw,
    int lda, int ldb, int m0, int n0, int k0, int tid,
    size_t maxA4, size_t maxB4, int M, int N, StageRegs& r)
{
    #pragma unroll
    for (int i = 0; i < 4; i++) {
        int e = tid + i * 256;
        int row = e >> 3, c4 = (e & 7) * 4;
        int gm = m0 + row;
        size_t idx = (gm < M) ? ((size_t)gm * lda + k0 + c4) : 0;
        idx = idx > maxA4 ? maxA4 : idx;
        r.a[i] = *(const float4*)(A + idx);
    }
    #pragma unroll
    for (int i = 0; i < 2; i++) {
        int e = tid + i * 256;
        int row = e >> 3, c4 = (e & 7) * 4;
        int gn = n0 + row;
        size_t idx = (gn < N) ? ((size_t)gn * ldb + k0 + c4) : 0;
        idx = idx > maxB4 ? maxB4 : idx;
        r.b[i] = *(const float4*)(Bw + idx);
    }
}

// Convert staged registers to bf16 (zeroing out-of-range elements) and commit
// to LDS as 8-byte stores.
__device__ __forceinline__ void stage_commit(
    const StageRegs& r, int m0, int n0, int k0, int tid,
    int M, int N, int K, __bf16* sA, __bf16* sB)
{
    #pragma unroll
    for (int i = 0; i < 4; i++) {
        int e = tid + i * 256;
        int row = e >> 3, c4 = (e & 7) * 4;
        int gm = m0 + row, gk = k0 + c4;
        bool rok = gm < M;
        float4 v = r.a[i];
        v4bf p = { (__bf16)((rok && gk + 0 < K) ? v.x : 0.0f),
                   (__bf16)((rok && gk + 1 < K) ? v.y : 0.0f),
                   (__bf16)((rok && gk + 2 < K) ? v.z : 0.0f),
                   (__bf16)((rok && gk + 3 < K) ? v.w : 0.0f) };
        *(v4bf*)&sA[row * LSTR + c4] = p;
    }
    #pragma unroll
    for (int i = 0; i < 2; i++) {
        int e = tid + i * 256;
        int row = e >> 3, c4 = (e & 7) * 4;
        int gn = n0 + row, gk = k0 + c4;
        bool rok = gn < N;
        float4 v = r.b[i];
        v4bf p = { (__bf16)((rok && gk + 0 < K) ? v.x : 0.0f),
                   (__bf16)((rok && gk + 1 < K) ? v.y : 0.0f),
                   (__bf16)((rok && gk + 2 < K) ? v.z : 0.0f),
                   (__bf16)((rok && gk + 3 < K) ? v.w : 0.0f) };
        *(v4bf*)&sB[row * LSTR + c4] = p;
    }
}

__global__ __launch_bounds__(256)
void gemm_bf16_wmma(const float* __restrict__ A, int lda,
                    const float* __restrict__ Bw, int ldb,
                    float* __restrict__ C, int ldc,
                    int M, int N, int K,
                    const float* __restrict__ bias)
{
    __shared__ __align__(16) __bf16 sA[TM * LSTR];
    __shared__ __align__(16) __bf16 sB[TN * LSTR];

    const int tid  = threadIdx.x;
    const int m0   = blockIdx.y * TM;
    const int n0   = blockIdx.x * TN;
    const int wid  = tid >> 5;
    const int lane = tid & 31;
    const int wm   = (wid >> 1) * 32;   // 4 wave-rows
    const int wn   = (wid & 1) * 32;    // 2 wave-cols
    const int lr   = lane & 15;
    const int hi   = lane >> 4;

    const size_t maxA4 = (size_t)M * lda - 4;
    const size_t maxB4 = (size_t)N * ldb - 4;

    v8f acc[2][2];
    for (int i = 0; i < 2; i++)
        for (int j = 0; j < 2; j++)
            acc[i][j] = v8f{};

    StageRegs cur, nxt;
    stage_issue(A, Bw, lda, ldb, m0, n0, 0, tid, maxA4, maxB4, M, N, cur);
    nxt = cur;

    for (int k0 = 0; k0 < K; k0 += TK) {
        stage_commit(cur, m0, n0, k0, tid, M, N, K, sA, sB);
        __syncthreads();

        // kick off next tile's global loads; they complete while WMMA runs
        if (k0 + TK < K)
            stage_issue(A, Bw, lda, ldb, m0, n0, k0 + TK, tid,
                        maxA4, maxB4, M, N, nxt);

        #pragma unroll
        for (int sm = 0; sm < 2; sm++) {
            // A fragment: lane lr = row M, hi selects K interleave
            // (V0-3: K = hi*8 + 0..7, V4-7: K = 16 + hi*8 + 0..7)
            int aoff = (wm + sm * 16 + lr) * LSTR + hi * 8;
            v8bf alo = *(const v8bf*)&sA[aoff];
            v8bf ahi = *(const v8bf*)&sA[aoff + 16];
            v16bf afrag = __builtin_shufflevector(alo, ahi,
                0,1,2,3,4,5,6,7,8,9,10,11,12,13,14,15);
            #pragma unroll
            for (int sn = 0; sn < 2; sn++) {
                // B fragment: lane lr = col N, hi selects K interleave (same as A)
                int boff = (wn + sn * 16 + lr) * LSTR + hi * 8;
                v8bf blo = *(const v8bf*)&sB[boff];
                v8bf bhi = *(const v8bf*)&sB[boff + 16];
                v16bf bfrag = __builtin_shufflevector(blo, bhi,
                    0,1,2,3,4,5,6,7,8,9,10,11,12,13,14,15);
                acc[sm][sn] = __builtin_amdgcn_wmma_f32_16x16x32_bf16(
                    false, afrag, false, bfrag, (short)0, acc[sm][sn],
                    false, false);
            }
        }
        __syncthreads();
        cur = nxt;
    }

    // epilogue: C/D layout — lane lr = col N, VGPR v = row (v + hi*8)
    #pragma unroll
    for (int sm = 0; sm < 2; sm++) {
        #pragma unroll
        for (int sn = 0; sn < 2; sn++) {
            int n = n0 + wn + sn * 16 + lr;
            if (n >= N) continue;
            float bv = bias ? bias[n] : 0.0f;
            #pragma unroll
            for (int v = 0; v < 8; v++) {
                int m = m0 + wm + sm * 16 + hi * 8 + v;
                if (m < M) C[(size_t)m * ldc + n] = acc[sm][sn][v] + bv;
            }
        }
    }
}

// ---- depthwise conv1d (k=4, pad (1,2)) + bias + SiLU for x and z ----------
__global__ __launch_bounds__(256)
void conv_silu_kernel(const float* __restrict__ xz,
                      const float* __restrict__ cxw, const float* __restrict__ cxb,
                      const float* __restrict__ czw, const float* __restrict__ czb,
                      float* __restrict__ x_act, float* __restrict__ yz)
{
    int id = blockIdx.x * 256 + threadIdx.x;
    if (id >= ROWS * DHALF) return;
    int row = id / DHALF;          // b*SEQ + l
    int d   = id - row * DHALF;
    int l   = row & (SEQ - 1);

    float ax = 0.0f, az = 0.0f;
    #pragma unroll
    for (int j = 0; j < 4; j++) {
        int ls = l - 1 + j;        // out[l] = sum_j w[j] * in[l-1+j]
        bool ok = (ls >= 0 && ls < SEQ);
        const float* p = xz + (size_t)(row + (ok ? ls - l : 0)) * DINNER;
        float vx = p[d], vz = p[DHALF + d];
        ax += cxw[d * 4 + j] * (ok ? vx : 0.0f);
        az += czw[d * 4 + j] * (ok ? vz : 0.0f);
    }
    ax += cxb[d];
    az += czb[d];
    float sx = ax / (1.0f + __expf(-ax));
    float sz = az / (1.0f + __expf(-az));
    x_act[(size_t)row * DHALF + d] = sx;
    yz[(size_t)row * DINNER + DHALF + d] = sz;   // z -> upper half of concat
}

// ---- selective scan, pass 1: per-chunk local scan with h_in = 0 -----------
// Writes partial y, plus per-chunk carry state h_end[16] and delta-sum.
__global__ __launch_bounds__(256)
void scan_chunk_kernel(const float* __restrict__ x_act,
                       const float* __restrict__ dt_full,
                       const float* __restrict__ x_dbl,
                       const float* __restrict__ A_log,
                       const float* __restrict__ D_param,
                       const float* __restrict__ b_dt,
                       float* __restrict__ yz,
                       float* __restrict__ hend,
                       float* __restrict__ Ssum)
{
    __shared__ float sBm[CHUNK][DSTATE];
    __shared__ float sCm[CHUNK][DSTATE];

    const int tid = threadIdx.x;
    const int b   = blockIdx.y;
    const int c   = blockIdx.z;
    const int d   = blockIdx.x * 256 + tid;
    const size_t row0 = (size_t)b * SEQ + c * CHUNK;

    // stage B (cols 48..63) and C (cols 64..79) of x_dbl for this chunk
    for (int i = tid; i < CHUNK * 2 * DSTATE; i += 256) {
        int ll = i >> 5, cc = i & 31;
        float v = x_dbl[(row0 + ll) * 80 + DTRANK + cc];
        if (cc < DSTATE) sBm[ll][cc] = v;
        else             sCm[ll][cc - DSTATE] = v;
    }
    __syncthreads();

    float h[DSTATE], An[DSTATE];
    #pragma unroll
    for (int n = 0; n < DSTATE; n++) {
        h[n]  = 0.0f;
        An[n] = -__expf(A_log[d * DSTATE + n]);
    }
    const float Dd  = D_param[d];
    const float bdt = b_dt[d];      // added AGAIN inside scan (reference quirk)

    float P = 0.0f;                 // running delta sum for this chunk
    for (int ll = 0; ll < CHUNK; ll++) {
        size_t rl = row0 + ll;
        float u     = x_act[rl * DHALF + d];
        float delta = softplus_f(dt_full[rl * DHALF + d] + bdt);
        P += delta;
        float du = delta * u;
        float y  = 0.0f;
        #pragma unroll
        for (int n = 0; n < DSTATE; n++) {
            float dA = __expf(delta * An[n]);
            h[n] = dA * h[n] + du * sBm[ll][n];
            y += h[n] * sCm[ll][n];
        }
        yz[rl * DINNER + d] = y + u * Dd;   // partial y (h_in term added later)
    }

    const size_t cid = (size_t)(b * NCH + c) * DHALF + d;
    #pragma unroll
    for (int n = 0; n < DSTATE; n++) hend[cid * DSTATE + n] = h[n];
    Ssum[cid] = P;
}

// ---- selective scan, pass 2: serial carry over the 16 chunks --------------
__global__ __launch_bounds__(256)
void scan_carry_kernel(const float* __restrict__ A_log,
                       const float* __restrict__ hend,
                       const float* __restrict__ Ssum,
                       float* __restrict__ hin_buf)
{
    const int tid = threadIdx.x;
    const int b   = blockIdx.y;
    const int d   = blockIdx.x * 256 + tid;

    float hin[DSTATE], An[DSTATE];
    #pragma unroll
    for (int n = 0; n < DSTATE; n++) {
        hin[n] = 0.0f;
        An[n]  = -__expf(A_log[d * DSTATE + n]);
    }
    for (int c = 0; c < NCH; c++) {
        const size_t cid = (size_t)(b * NCH + c) * DHALF + d;
        #pragma unroll
        for (int n = 0; n < DSTATE; n++) hin_buf[cid * DSTATE + n] = hin[n];
        float S = Ssum[cid];
        #pragma unroll
        for (int n = 0; n < DSTATE; n++)
            hin[n] = __expf(An[n] * S) * hin[n] + hend[cid * DSTATE + n];
    }
}

// ---- selective scan, pass 3: add C_t . (h_in * exp(A * prefix)) -----------
__global__ __launch_bounds__(256)
void scan_fixup_kernel(const float* __restrict__ dt_full,
                       const float* __restrict__ x_dbl,
                       const float* __restrict__ A_log,
                       const float* __restrict__ b_dt,
                       const float* __restrict__ hin_buf,
                       float* __restrict__ yz)
{
    __shared__ float sCm[CHUNK][DSTATE];

    const int tid = threadIdx.x;
    const int b   = blockIdx.y;
    const int c   = blockIdx.z + 1;           // chunk 0 has h_in = 0
    const int d   = blockIdx.x * 256 + tid;
    const size_t row0 = (size_t)b * SEQ + c * CHUNK;

    for (int i = tid; i < CHUNK * DSTATE; i += 256) {
        int ll = i >> 4, cc = i & 15;
        sCm[ll][cc] = x_dbl[(row0 + ll) * 80 + DTRANK + DSTATE + cc];
    }
    __syncthreads();

    const size_t cid = (size_t)(b * NCH + c) * DHALF + d;
    float hin[DSTATE], An[DSTATE];
    #pragma unroll
    for (int n = 0; n < DSTATE; n++) {
        hin[n] = hin_buf[cid * DSTATE + n];
        An[n]  = -__expf(A_log[d * DSTATE + n]);
    }
    const float bdt = b_dt[d];

    float P = 0.0f;
    for (int ll = 0; ll < CHUNK; ll++) {
        size_t rl = row0 + ll;
        float delta = softplus_f(dt_full[rl * DHALF + d] + bdt);
        P += delta;                      // inclusive prefix
        float y = 0.0f;
        #pragma unroll
        for (int n = 0; n < DSTATE; n++)
            y += sCm[ll][n] * hin[n] * __expf(An[n] * P);
        yz[rl * DINNER + d] += y;
    }
}

// ---------------------------------------------------------------------------
extern "C" void kernel_launch(void* const* d_in, const int* in_sizes, int n_in,
                              void* d_out, int out_size, void* d_ws, size_t ws_size,
                              hipStream_t stream)
{
    const float* hidden = (const float*)d_in[0];   // [8,1024,768]
    const float* W_in   = (const float*)d_in[1];   // [1536,768]
    const float* cxw    = (const float*)d_in[2];   // [768,1,4]
    const float* cxb    = (const float*)d_in[3];   // [768]
    const float* czw    = (const float*)d_in[4];   // [768,1,4]
    const float* czb    = (const float*)d_in[5];   // [768]
    const float* W_x    = (const float*)d_in[6];   // [80,768]
    const float* W_dt   = (const float*)d_in[7];   // [768,48]
    const float* b_dt   = (const float*)d_in[8];   // [768]
    const float* A_log  = (const float*)d_in[9];   // [768,16]
    const float* D_par  = (const float*)d_in[10];  // [768]
    const float* W_out  = (const float*)d_in[11];  // [768,1536]
    float* out = (float*)d_out;                    // [8,1024,768] f32

    // workspace carve-up (floats)
    float* xz      = (float*)d_ws;                        // [8192,1536]
    float* x_act   = xz + (size_t)ROWS * DINNER;          // [8192,768]
    float* yz      = x_act + (size_t)ROWS * DHALF;        // [8192,1536] = [y|z]
    float* x_dbl   = yz + (size_t)ROWS * DINNER;          // [8192,80]
    float* hend    = x_dbl + (size_t)ROWS * 80;           // [8,16,768,16]
    float* hin_buf = hend + (size_t)BSZ * NCH * DHALF * DSTATE;
    float* Ssum    = hin_buf + (size_t)BSZ * NCH * DHALF * DSTATE; // [8,16,768]
    float* dt_full = xz;  // xz is dead after conv; reuse for dt  [8192,768]

    dim3 blk(256);

    // 1. xz = hidden @ W_in^T                (M=8192, N=1536, K=768)
    gemm_bf16_wmma<<<dim3(DINNER / TN, ROWS / TM), blk, 0, stream>>>(
        hidden, DMODEL, W_in, DMODEL, xz, DINNER, ROWS, DINNER, DMODEL, nullptr);

    // 2. depthwise conv + SiLU -> x_act, and z half of yz
    conv_silu_kernel<<<(ROWS * DHALF + 255) / 256, blk, 0, stream>>>(
        xz, cxw, cxb, czw, czb, x_act, yz);

    // 3. x_dbl = x_act @ W_x^T               (N=80, guarded)
    gemm_bf16_wmma<<<dim3((80 + TN - 1) / TN, ROWS / TM), blk, 0, stream>>>(
        x_act, DHALF, W_x, DHALF, x_dbl, 80, ROWS, 80, DHALF, nullptr);

    // 4. dt_full = x_dbl[:, :48] @ W_dt^T + b_dt   (K=48, zero-padded)
    gemm_bf16_wmma<<<dim3(DHALF / TN, ROWS / TM), blk, 0, stream>>>(
        x_dbl, 80, W_dt, DTRANK, dt_full, DHALF, ROWS, DHALF, DTRANK, b_dt);

    // 5a. chunked local scans (16x parallelism over sequence)
    scan_chunk_kernel<<<dim3(DHALF / 256, BSZ, NCH), blk, 0, stream>>>(
        x_act, dt_full, x_dbl, A_log, D_par, b_dt, yz, hend, Ssum);

    // 5b. serial carry scan across chunks (tiny)
    scan_carry_kernel<<<dim3(DHALF / 256, BSZ), blk, 0, stream>>>(
        A_log, hend, Ssum, hin_buf);

    // 5c. fixup: add incoming-state contribution for chunks 1..15
    scan_fixup_kernel<<<dim3(DHALF / 256, BSZ, NCH - 1), blk, 0, stream>>>(
        dt_full, x_dbl, A_log, b_dt, hin_buf, yz);

    // 6. out = [y|z] @ W_out^T               (M=8192, N=768, K=1536)
    gemm_bf16_wmma<<<dim3(DMODEL / TN, ROWS / TM), blk, 0, stream>>>(
        yz, DINNER, W_out, DINNER, out, DMODEL, ROWS, DMODEL, DINNER, nullptr);
}
